// StoryMemoryBlock_64201171140752
// MI455X (gfx1250) — matrix operand order repaired and checked by reading, not compiled, not run on hardware
//
#include <hip/hip_runtime.h>
#include <hip/hip_bf16.h>

// ---------------- types ----------------
typedef __attribute__((ext_vector_type(16))) __bf16 v16bf;
typedef __attribute__((ext_vector_type(8)))  float  v8f;
typedef __attribute__((ext_vector_type(4)))  int    v4i;

union FragBF { v16bf v; uint4 q[2]; };

static __device__ __forceinline__ __bf16 f2bf(float f) {
    unsigned u = __float_as_uint(f);
    unsigned r = (u + 0x7FFFu + ((u >> 16) & 1u)) >> 16;
    unsigned short h = (unsigned short)r;
    return __builtin_bit_cast(__bf16, h);
}
static __device__ __forceinline__ float bf2f(__bf16 b) {
    unsigned short h = __builtin_bit_cast(unsigned short, b);
    return __uint_as_float(((unsigned)h) << 16);
}
static __device__ __forceinline__ float sigmoidf_(float x) { return 1.f / (1.f + __expf(-x)); }

// ---- CDNA5 async global->LDS copy (ASYNCcnt-tracked), with inline-asm fallback ----
static __device__ __forceinline__ void async_ld_b128(__bf16* lds, const __bf16* g) {
#if __has_builtin(__builtin_amdgcn_global_load_async_to_lds_b128)
    __builtin_amdgcn_global_load_async_to_lds_b128(
        (__attribute__((address_space(1))) v4i*)(g),
        (__attribute__((address_space(3))) v4i*)(lds), 0, 0);
#else
    unsigned laddr = (unsigned)(unsigned long long)(__attribute__((address_space(3))) char*)lds;
    asm volatile("global_load_async_to_lds_b128 %0, %1, off" :: "v"(laddr), "v"(g) : "memory");
#endif
}
#if __has_builtin(__builtin_amdgcn_s_wait_asynccnt)
#define WAIT_ASYNC(n) __builtin_amdgcn_s_wait_asynccnt(n)
#else
#define WAIT_ASYNC(n) asm volatile("s_wait_asynccnt " #n ::: "memory")
#endif

// ---------------- problem constants ----------------
#define BB 4
#define TT 4096
#define DD 1024
#define DFF 4096
#define KK 15
#define SS 8
#define SDD 32
#define MM (BB * TT)          // 16384 rows

// ---------------- elementwise: fp32 -> bf16 cast ----------------
__global__ __launch_bounds__(256) void cast_bf16_kernel(const float* __restrict__ src,
                                                        __bf16* __restrict__ dst, size_t n) {
    size_t i = (size_t)blockIdx.x * 256 + threadIdx.x;
    if (i < n) dst[i] = f2bf(src[i]);
}

// ---------------- rmsnorm (fp32 in, bf16 out), one block per row ----------------
__global__ __launch_bounds__(256) void rmsnorm_bf16_kernel(const float* __restrict__ x,
                                                           const float* __restrict__ w,
                                                           __bf16* __restrict__ out) {
    __shared__ float red[256];
    size_t row = blockIdx.x;
    const float* xr = x + row * DD;
    float ss = 0.f;
    for (int i = threadIdx.x; i < DD; i += 256) { float v = xr[i]; ss += v * v; }
    red[threadIdx.x] = ss;
    __syncthreads();
    for (int s = 128; s > 0; s >>= 1) {
        if (threadIdx.x < s) red[threadIdx.x] += red[threadIdx.x + s];
        __syncthreads();
    }
    float scale = rsqrtf(red[0] / (float)DD + 1e-6f);
    __bf16* orow = out + row * DD;
    for (int i = threadIdx.x; i < DD; i += 256) orow[i] = f2bf(xr[i] * scale * w[i]);
}

// ---------------- WMMA bf16 GEMM:  C[M,N] = A[M,K] @ W[N,K]^T  ----------------
// Block tile 256(M) x 64(N), 8 waves, each wave = 32(M) x 64(N), K-step 32.
// Double-buffered LDS fed by async global->LDS (5 async b128 per thread per tile).
#define BM 256
#define BN 64
#define LDA 40   // padded LDS stride (elems) to avoid bank conflicts
#define LDB 40
enum { EPI_STORE = 0, EPI_ADDX = 1, EPI_SWIGLU = 2 };

template <int EPI>
__global__ __launch_bounds__(256) void wmma_gemm_kernel(const __bf16* __restrict__ A,
                                                        const __bf16* __restrict__ W,
                                                        float* __restrict__ Cf,
                                                        __bf16* __restrict__ Cb,
                                                        const float* __restrict__ aux,
                                                        int M, int N, int K) {
    __shared__ __align__(16) __bf16 ldsA[2][BM * LDA];
    __shared__ __align__(16) __bf16 ldsB[2][BN * LDB];

    const int row0 = blockIdx.y * BM;
    const int n0   = blockIdx.x * BN;
    const int lane = threadIdx.x & 31;
    const int wm   = threadIdx.x >> 5;          // wave id: 32-row M sub-slab
    const int lrow = lane & 15;
    const int lhi  = lane >> 4;

    v8f acc[2][4];
#pragma unroll
    for (int mi = 0; mi < 2; ++mi)
#pragma unroll
        for (int s = 0; s < 4; ++s)
#pragma unroll
            for (int j = 0; j < 8; ++j) acc[mi][s][j] = 0.f;

    const int rA = threadIdx.x;                 // A coop: one 32-elem row per thread
    const int rB = threadIdx.x >> 2, qB = threadIdx.x & 3;

    const int niter = K / 32;
    auto issue = [&](int it) {                  // async-stage tile `it` into buffer it&1
        const int k0 = it * 32;
        const int buf = it & 1;
        const __bf16* ga = A + (size_t)(row0 + rA) * K + k0;
        __bf16* la = &ldsA[buf][rA * LDA];
        async_ld_b128(la + 0,  ga + 0);
        async_ld_b128(la + 8,  ga + 8);
        async_ld_b128(la + 16, ga + 16);
        async_ld_b128(la + 24, ga + 24);
        async_ld_b128(&ldsB[buf][rB * LDB + qB * 8],
                      W + (size_t)(n0 + rB) * K + k0 + qB * 8);
        if (k0 + 64 < K) {                      // pull tile it+2 toward L2
            __builtin_prefetch(ga + 64, 0, 0);
        }
    };

    issue(0);
    for (int it = 0; it < niter; ++it) {
        const int buf = it & 1;
        if (it + 1 < niter) {
            issue(it + 1);
            WAIT_ASYNC(5);                      // in-order: tile `it`'s 5 loads are done
        } else {
            WAIT_ASYNC(0);
        }
        __syncthreads();                        // tile `it` visible to all waves

        // A fragments: lane<16 holds row lrow K[0..7],K[16..23]; lane>=16 K[8..15],K[24..31]
        FragBF af[2];
#pragma unroll
        for (int mi = 0; mi < 2; ++mi) {
            const __bf16* arow = &ldsA[buf][(wm * 32 + mi * 16 + lrow) * LDA];
            af[mi].q[0] = *(const uint4*)(arow + lhi * 8);
            af[mi].q[1] = *(const uint4*)(arow + lhi * 8 + 16);
        }
        // B fragments: lane<16 holds column (sub*16 + lrow) K[0..7],K[16..23]; lane>=16 K[8..15],K[24..31]
        FragBF bf[4];
#pragma unroll
        for (int sub = 0; sub < 4; ++sub) {
            const __bf16* brow = &ldsB[buf][(sub * 16 + lrow) * LDB + lhi * 8];
            bf[sub].q[0] = *(const uint4*)(brow);
            bf[sub].q[1] = *(const uint4*)(brow + 16);
        }
#pragma unroll
        for (int mi = 0; mi < 2; ++mi)
#pragma unroll
            for (int sub = 0; sub < 4; ++sub)
                acc[mi][sub] = __builtin_amdgcn_wmma_f32_16x16x32_bf16(
                    false, af[mi].v, false, bf[sub].v, (short)0, acc[mi][sub], false, false);

        __syncthreads();                        // everyone done reading buf before reuse
    }

    // epilogue: C/D layout -> element (m = j + 8*lhi, n = lrow) of each 16x16 tile
#pragma unroll
    for (int mi = 0; mi < 2; ++mi) {
        const int rbase = row0 + wm * 32 + mi * 16 + 8 * lhi;
#pragma unroll
        for (int sub = 0; sub < 4; ++sub) {
            const int ncol = n0 + sub * 16 + lrow;
#pragma unroll
            for (int j = 0; j < 8; ++j) {
                size_t idx = (size_t)(rbase + j) * N + ncol;
                float v = acc[mi][sub][j];
                if (EPI == EPI_STORE) {
                    Cf[idx] = v;
                } else if (EPI == EPI_ADDX) {
                    Cf[idx] = aux[idx] + v;
                } else {  // EPI_SWIGLU: Cb = silu(aux) * v
                    float g = aux[idx];
                    Cb[idx] = f2bf(g * sigmoidf_(g) * v);
                }
            }
        }
    }
}

// ---------------- fused causal depthwise conv (K=15) * sigmoid(gate) -> bf16 ----------------
// gv: (B*T, 2D) fp32; gate = gv[:, :D], val = gv[:, D:2D]
__global__ __launch_bounds__(256) void conv_gate_kernel(const float* __restrict__ gv,
                                                        const float* __restrict__ conv_w,
                                                        __bf16* __restrict__ out) {
    int d = blockIdx.x * 256 + threadIdx.x;
    int t = blockIdx.y;
    int b = blockIdx.z;
    size_t bt = (size_t)b * TT + t;
    float accv = 0.f;
#pragma unroll
    for (int k = 0; k < KK; ++k) {
        int tt = t - (KK - 1) + k;
        if (tt >= 0) {
            accv += conv_w[d * KK + k] * gv[((size_t)b * TT + tt) * (2 * DD) + DD + d];
        }
    }
    float graw = gv[bt * (2 * DD) + d];
    out[bt * DD + d] = f2bf(accv * sigmoidf_(graw));
}

// ---------------- memory projections: write_g(sig), write_v, read_w(softmax) ----------------
__global__ __launch_bounds__(256) void mem_proj_kernel(const __bf16* __restrict__ h,
                                                       const float* __restrict__ wgate,
                                                       const float* __restrict__ wval,
                                                       const float* __restrict__ rquery,
                                                       float* __restrict__ write_g,
                                                       float* __restrict__ write_v,
                                                       float* __restrict__ read_w) {
    __shared__ float hs[DD];
    __shared__ float outs[SS + SDD + SS];  // 48
    size_t row = blockIdx.x;
    for (int i = threadIdx.x; i < DD; i += 256) hs[i] = bf2f(h[row * DD + i]);
    __syncthreads();
    int o = threadIdx.x;
    if (o < SS + SDD + SS) {
        const float* w = (o < SS) ? (wgate + o * DD)
                       : (o < SS + SDD) ? (wval + (o - SS) * DD)
                                        : (rquery + (o - SS - SDD) * DD);
        float acc = 0.f;
        for (int i = 0; i < DD; ++i) acc += hs[i] * w[i];
        outs[o] = acc;
    }
    __syncthreads();
    if (threadIdx.x < SS) write_g[row * SS + threadIdx.x] = sigmoidf_(outs[threadIdx.x]);
    if (threadIdx.x < SDD) write_v[row * SDD + threadIdx.x] = outs[SS + threadIdx.x];
    if (threadIdx.x == 0) {
        float mx = outs[SS + SDD];
        for (int j = 1; j < SS; ++j) mx = fmaxf(mx, outs[SS + SDD + j]);
        float e[SS], sum = 0.f;
        for (int j = 0; j < SS; ++j) { e[j] = __expf(outs[SS + SDD + j] - mx); sum += e[j]; }
        float inv = 1.f / sum;
        for (int j = 0; j < SS; ++j) read_w[row * SS + j] = e[j] * inv;
    }
}

// ---------------- gated linear recurrence over slots (sequential in T) ----------------
__global__ __launch_bounds__(256) void scan_kernel(const float* __restrict__ mem0,
                                                   const float* __restrict__ wg,
                                                   const float* __restrict__ wv,
                                                   float* __restrict__ stack,
                                                   float* __restrict__ cur_out) {
    int b = blockIdx.x;
    int s = threadIdx.x >> 5;   // slot
    int e = threadIdx.x & 31;   // element
    float cur = mem0[((size_t)b * SS + s) * SDD + e];
    for (int t = 0; t < TT; ++t) {
        size_t bt = (size_t)b * TT + t;
        float w = wg[bt * SS + s];
        float v = wv[bt * SDD + e];
        cur = w * v + (1.f - w) * cur;
        stack[bt * (SS * SDD) + threadIdx.x] = cur;
    }
    cur_out[((size_t)b * SS + s) * SDD + e] = cur;
}

// ---------------- read_flat = read_w * memory_stack -> bf16 ----------------
__global__ __launch_bounds__(256) void read_flat_kernel(const float* __restrict__ stack,
                                                        const float* __restrict__ rw,
                                                        __bf16* __restrict__ rf, size_t n) {
    size_t i = (size_t)blockIdx.x * 256 + threadIdx.x;
    if (i >= n) return;
    size_t bt = i >> 8;
    int s = (int)((i >> 5) & 7);
    rf[i] = f2bf(rw[bt * SS + s] * stack[i]);
}

// =================================================================================
extern "C" void kernel_launch(void* const* d_in, const int* in_sizes, int n_in,
                              void* d_out, int out_size, void* d_ws, size_t ws_size,
                              hipStream_t stream) {
    const float* x_in    = (const float*)d_in[0];   // (B,T,D)
    const float* memory  = (const float*)d_in[1];   // (B,S,SD)
    const float* ln1_w   = (const float*)d_in[2];
    const float* up_w    = (const float*)d_in[3];   // (2D,D)
    const float* conv_w  = (const float*)d_in[4];   // (D,1,K)
    const float* down_w  = (const float*)d_in[5];   // (D,D)
    const float* lnm_w   = (const float*)d_in[6];
    const float* wgate   = (const float*)d_in[7];   // (S,D)
    const float* wval    = (const float*)d_in[8];   // (SD,D)
    const float* rquery  = (const float*)d_in[9];   // (S,D)
    const float* rout_w  = (const float*)d_in[10];  // (D,S*SD)
    const float* ln2_w   = (const float*)d_in[11];
    const float* Wg_w    = (const float*)d_in[12];  // (DF,D)
    const float* Wu_w    = (const float*)d_in[13];  // (DF,D)
    const float* Wo_w    = (const float*)d_in[14];  // (D,DF)
    float* out = (float*)d_out;                     // x (M*D) then cur (B*S*SD)

    const size_t M = MM;
    char* base = (char*)d_ws;
    size_t off = 0;
    auto alloc = [&](size_t bytes) { size_t r = off; off += (bytes + 255) & ~(size_t)255; return r; };

    __bf16* h_bf   = (__bf16*)(base + alloc(M * DD * 2));
    __bf16* wup_b  = (__bf16*)(base + alloc((size_t)2 * DD * DD * 2));
    __bf16* wdn_b  = (__bf16*)(base + alloc((size_t)DD * DD * 2));
    __bf16* wro_b  = (__bf16*)(base + alloc((size_t)DD * SS * SDD * 2));
    __bf16* wgg_b  = (__bf16*)(base + alloc((size_t)DFF * DD * 2));
    __bf16* wuu_b  = (__bf16*)(base + alloc((size_t)DFF * DD * 2));
    __bf16* woo_b  = (__bf16*)(base + alloc((size_t)DD * DFF * 2));
    float*  gv     = (float*)(base + alloc(M * 2 * DD * 4));   // reused by FFN chunks
    __bf16* md_bf  = (__bf16*)(base + alloc(M * DD * 2));
    float*  x_ws   = (float*)(base + alloc(M * DD * 4));
    float*  wg_buf = (float*)(base + alloc(M * SS * 4));
    float*  wv_buf = (float*)(base + alloc(M * SDD * 4));
    float*  rw_buf = (float*)(base + alloc(M * SS * 4));
    float*  stack  = (float*)(base + alloc(M * SS * SDD * 4));
    __bf16* rf_bf  = (__bf16*)(base + alloc(M * SS * SDD * 2));

    dim3 blk(256);

    // --- weight casts to bf16 ---
    auto cast = [&](const float* s, __bf16* d, size_t n) {
        cast_bf16_kernel<<<dim3((unsigned)((n + 255) / 256)), blk, 0, stream>>>(s, d, n);
    };
    cast(up_w,   wup_b, (size_t)2 * DD * DD);
    cast(down_w, wdn_b, (size_t)DD * DD);
    cast(rout_w, wro_b, (size_t)DD * SS * SDD);
    cast(Wg_w,   wgg_b, (size_t)DFF * DD);
    cast(Wu_w,   wuu_b, (size_t)DFF * DD);
    cast(Wo_w,   woo_b, (size_t)DD * DFF);

    // --- stage 1: mixer ---
    rmsnorm_bf16_kernel<<<dim3((unsigned)M), blk, 0, stream>>>(x_in, ln1_w, h_bf);
    wmma_gemm_kernel<EPI_STORE><<<dim3(2 * DD / BN, (unsigned)(M / BM)), blk, 0, stream>>>(
        h_bf, wup_b, gv, nullptr, nullptr, (int)M, 2 * DD, DD);
    conv_gate_kernel<<<dim3(DD / 256, TT, BB), blk, 0, stream>>>(gv, conv_w, md_bf);
    wmma_gemm_kernel<EPI_ADDX><<<dim3(DD / BN, (unsigned)(M / BM)), blk, 0, stream>>>(
        md_bf, wdn_b, x_ws, nullptr, x_in, (int)M, DD, DD);

    // --- stage 2: story memory ---
    rmsnorm_bf16_kernel<<<dim3((unsigned)M), blk, 0, stream>>>(x_ws, lnm_w, h_bf);
    mem_proj_kernel<<<dim3((unsigned)M), blk, 0, stream>>>(h_bf, wgate, wval, rquery,
                                                           wg_buf, wv_buf, rw_buf);
    scan_kernel<<<dim3(BB), blk, 0, stream>>>(memory, wg_buf, wv_buf, stack,
                                              out + M * DD);   // cur -> output tail
    read_flat_kernel<<<dim3((unsigned)(M * SS * SDD / 256)), blk, 0, stream>>>(
        stack, rw_buf, rf_bf, M * SS * SDD);
    wmma_gemm_kernel<EPI_ADDX><<<dim3(DD / BN, (unsigned)(M / BM)), blk, 0, stream>>>(
        rf_bf, wro_b, x_ws, nullptr, x_ws, (int)M, DD, SS * SDD);

    // --- stage 3: SwiGLU FFN, chunked over M to reuse the gv buffer ---
    rmsnorm_bf16_kernel<<<dim3((unsigned)M), blk, 0, stream>>>(x_ws, ln2_w, h_bf);
    const size_t Mc = 4096;
    float*  gbuf  = gv;                                  // Mc*DF fp32 = 64MB
    __bf16* ffbuf = (__bf16*)((char*)gv + Mc * DFF * 4); // Mc*DF bf16 = 32MB (region is 128MB)
    for (size_t c = 0; c < M / Mc; ++c) {
        const size_t r0 = c * Mc;
        wmma_gemm_kernel<EPI_STORE><<<dim3(DFF / BN, (unsigned)(Mc / BM)), blk, 0, stream>>>(
            h_bf + r0 * DD, wgg_b, gbuf, nullptr, nullptr, (int)Mc, DFF, DD);
        wmma_gemm_kernel<EPI_SWIGLU><<<dim3(DFF / BN, (unsigned)(Mc / BM)), blk, 0, stream>>>(
            h_bf + r0 * DD, wuu_b, nullptr, ffbuf, gbuf, (int)Mc, DFF, DD);
        wmma_gemm_kernel<EPI_ADDX><<<dim3(DD / BN, (unsigned)(Mc / BM)), blk, 0, stream>>>(
            ffbuf, woo_b, out + r0 * DD, nullptr, x_ws + r0 * DD, (int)Mc, DD, DFF);
    }
}
